// GCN_64811056497144
// MI455X (gfx1250) — matrix operand order repaired and checked by reading, not compiled, not run on hardware
//
#include <hip/hip_runtime.h>

// ---------------------------------------------------------------------------
// GCN for MI455X (gfx1250): WMMA bf16 GEMMs (fragment-packed operands)
//                           + f32-atomic edge aggregation
// ---------------------------------------------------------------------------

typedef __attribute__((ext_vector_type(16))) __bf16 v16bf;
typedef __attribute__((ext_vector_type(8)))  float  v8f;

#define N_NODES  100000
#define N_EDGES  1600000
#define D_IN     128
#define HID      128
#define H2       64
#define N_GR     128
#define N_CLS    10

// ---- WMMA fragment layout helpers (CDNA5 ISA 7.12.2, wave32) --------------
//
// A (16x32 bf16): lanes 0-15 row M=lane, K{0..7}+{16..23}; lanes 16-31 same M,
// K{8..15}+{24..31}.  Packed LDS layout: [kb][lane][16 bf16] so a fragment is
// one contiguous 32B-per-lane load (2x ds_load_b128).
//
// B (32x16 bf16): lane = col N (replicated in both halves), lanes 0-15 K{0..7}+{16..23},
// lanes 16-31 K{8..15}+{24..31}, element jj -> k = kbase + jj.  Weights are pre-packed
// in global memory: [kb][nt][lane][16 bf16] -> 2x global_load_b128 / fragment.

// packed-A index for element (row, col) of a row-major 16x128 tile
static __device__ inline int packedA_index(int row, int col) {
  int kb = col >> 5;
  int kk = col & 31;
  int lane = row + ((kk & 8) ? 16 : 0);
  int j = ((kk & 16) ? 4 : 0) + ((kk & 7) >> 1);
  int jj = 2 * j + (kk & 1);
  return (kb * 32 + lane) * 16 + jj;
}

static __device__ inline v16bf frag_a_packed(const __bf16* t, int lane, int kb) {
  return *(const v16bf*)(t + (size_t)(kb * 32 + lane) * 16);
}

static __device__ inline v16bf frag_b_packed(const __bf16* __restrict__ P,
                                             int nt_count, int lane, int kb, int nt) {
  return *(const v16bf*)(P + (size_t)((kb * nt_count + nt) * 32 + lane) * 16);
}

static __device__ inline v8f wmma_bf16(v16bf a, v16bf b, v8f c) {
  return __builtin_amdgcn_wmma_f32_16x16x32_bf16(false, a, false, b, (short)0, c,
                                                 false, false);
}

// source index in a row-major [K][ldn] weight for packed element i
static __device__ inline int b_src_index(int i, int nt_count, int ldn) {
  int jj = i & 15;
  int lane = (i >> 4) & 31;
  int frag = i >> 9;
  int nt = frag % nt_count;
  int kb = frag / nt_count;
  int n = nt * 16 + (lane & 15);
  int kbase = kb * 32 + ((lane < 16) ? 0 : 8) + ((jj & 8) ? 8 : 0);
  return (kbase + jj) * ldn + n;
}

// ---- small utility kernels ------------------------------------------------

// deg starts at 1.0 (self loop); pool buffers zeroed every call.
__global__ void k_init(float* __restrict__ degdis, float* __restrict__ sums,
                       float* __restrict__ cnt) {
  int i = blockIdx.x * blockDim.x + threadIdx.x;
  if (i < N_NODES) degdis[i] = 1.0f;
  if (i < N_GR * H2) sums[i] = 0.0f;
  if (i < N_GR) cnt[i] = 0.0f;
}

__global__ void k_degree(const int* __restrict__ dst, float* __restrict__ deg) {
  int e = blockIdx.x * blockDim.x + threadIdx.x;
  if (e < N_EDGES) atomicAdd(&deg[dst[e]], 1.0f);
}

__global__ void k_dis(float* __restrict__ degdis) {
  int i = blockIdx.x * blockDim.x + threadIdx.x;
  if (i < N_NODES) {
    float d = degdis[i];
    degdis[i] = (d > 0.0f) ? rsqrtf(d) : 0.0f;
  }
}

// convert weights to bf16 AND shuffle into WMMA B-fragment-packed layout
__global__ void k_packw(const float* __restrict__ w1, const float* __restrict__ wc1,
                        const float* __restrict__ wc2, __bf16* __restrict__ p1,
                        __bf16* __restrict__ pc1, __bf16* __restrict__ pc2) {
  int i = blockIdx.x * blockDim.x + threadIdx.x;
  if (i < D_IN * D_IN) {
    int s = b_src_index(i, 8, 128);
    p1[i]  = (__bf16)w1[s];
    pc1[i] = (__bf16)wc1[s];
  }
  if (i < HID * H2) {
    pc2[i] = (__bf16)wc2[b_src_index(i, 4, 64)];
  }
}

// ---- fused lin1 + conv1-weight double GEMM (WMMA) -------------------------
// per wave: 16 rows.  tmp = x@W1 + b1 ; hs1 = dis * (tmp @ Wc1)
// acc1 initialized to hs1 (this IS the self-loop message contribution).
__global__ __launch_bounds__(128) void k_gemm1(
    const float* __restrict__ x, const float* __restrict__ lin1_b,
    const __bf16* __restrict__ W1, const __bf16* __restrict__ Wc1,
    const float* __restrict__ dis, __bf16* __restrict__ hs1,
    float* __restrict__ acc1) {
  __shared__ __align__(16) __bf16 ldsA[4][16 * 128];
  __shared__ __align__(16) __bf16 ldsT[4][16 * 128];
  const int wave = threadIdx.x >> 5;
  const int lane = threadIdx.x & 31;
  const int r0 = (blockIdx.x * 4 + wave) * 16;

  // stage 16x128 bf16 tile of x, written directly in packed-A layout
#pragma unroll 4
  for (int i = 0; i < 64; ++i) {
    int idx = i * 32 + lane;
    int row = idx >> 7, col = idx & 127;
    int gr = r0 + row;
    float v = (gr < N_NODES) ? x[(size_t)gr * 128 + col] : 0.0f;
    ldsA[wave][packedA_index(row, col)] = (__bf16)v;
  }
  __syncthreads();

  const v8f vz = {0.f, 0.f, 0.f, 0.f, 0.f, 0.f, 0.f, 0.f};
  v8f acc[8];
#pragma unroll
  for (int nt = 0; nt < 8; ++nt) acc[nt] = vz;

  // GEMM1: 16x128 @ 128x128
#pragma unroll
  for (int kb = 0; kb < 4; ++kb) {
    v16bf a = frag_a_packed(ldsA[wave], lane, kb);
#pragma unroll
    for (int nt = 0; nt < 8; ++nt) {
      v16bf b = frag_b_packed(W1, 8, lane, kb, nt);
      acc[nt] = wmma_bf16(a, b, acc[nt]);
    }
  }

  // tmp = acc + lin1_b -> bf16 LDS, written directly in packed-A layout
  // (D layout: lane = col within tile, vgpr v = row v / v+8)
  {
    const int ncol = lane & 15;
    const int mbase = (lane < 16) ? 0 : 8;
#pragma unroll
    for (int nt = 0; nt < 8; ++nt) {
      float bv = lin1_b[nt * 16 + ncol];
#pragma unroll
      for (int v = 0; v < 8; ++v)
        ldsT[wave][packedA_index(mbase + v, nt * 16 + ncol)] =
            (__bf16)(acc[nt][v] + bv);
    }
  }
  __syncthreads();

  // GEMM2: tmp @ Wc1
#pragma unroll
  for (int nt = 0; nt < 8; ++nt) acc[nt] = vz;
#pragma unroll
  for (int kb = 0; kb < 4; ++kb) {
    v16bf a = frag_a_packed(ldsT[wave], lane, kb);
#pragma unroll
    for (int nt = 0; nt < 8; ++nt) {
      v16bf b = frag_b_packed(Wc1, 8, lane, kb, nt);
      acc[nt] = wmma_bf16(a, b, acc[nt]);
    }
  }

  // hs1 = dis[row] * result (pre-scaled message); acc1 init = self-loop msg
  {
    const int ncol = lane & 15;
    const int mbase = (lane < 16) ? 0 : 8;
#pragma unroll
    for (int nt = 0; nt < 8; ++nt)
#pragma unroll
      for (int v = 0; v < 8; ++v) {
        int gr = r0 + mbase + v;
        if (gr < N_NODES) {
          float val = acc[nt][v] * dis[gr];
          size_t o = (size_t)gr * 128 + nt * 16 + ncol;
          hs1[o] = (__bf16)val;
          acc1[o] = val;
        }
      }
  }
}

// ---- edge scatter-add, 128 features (bf16 gather, f32 atomic add) ---------
__global__ void k_agg128(const int* __restrict__ src, const int* __restrict__ dst,
                         const __bf16* __restrict__ hs, float* __restrict__ acc) {
  long long t = (long long)blockIdx.x * blockDim.x + threadIdx.x;
  int e = (int)(t >> 7);
  int f = (int)(t & 127);
  if (e < N_EDGES) {
    int s = src[e], d = dst[e];
    atomicAdd(&acc[(size_t)d * 128 + f], (float)hs[(size_t)s * 128 + f]);
  }
}

// ---- fused conv1 epilogue (bias, dropout, relu) + conv2-weight GEMM -------
__global__ __launch_bounds__(128) void k_post1_gemm2(
    const float* __restrict__ acc1, const float* __restrict__ conv1_b,
    const float* __restrict__ drop_u, const __bf16* __restrict__ Wc2,
    const float* __restrict__ dis, __bf16* __restrict__ hs2,
    float* __restrict__ acc2) {
  __shared__ __align__(16) __bf16 ldsT[4][16 * 128];
  const int wave = threadIdx.x >> 5;
  const int lane = threadIdx.x & 31;
  const int r0 = (blockIdx.x * 4 + wave) * 16;

#pragma unroll 4
  for (int i = 0; i < 64; ++i) {
    int idx = i * 32 + lane;
    int row = idx >> 7, col = idx & 127;
    int gr = r0 + row;
    float h = 0.0f;
    if (gr < N_NODES) {
      size_t o = (size_t)gr * 128 + col;
      float v = dis[gr] * acc1[o] + conv1_b[col];   // conv1 output
      v *= (drop_u[o] >= 0.5f) ? 2.0f : 0.0f;       // dropout p=0.5, train
      h = fmaxf(v, 0.0f);                            // relu
    }
    ldsT[wave][packedA_index(row, col)] = (__bf16)h;
  }
  __syncthreads();

  const v8f vz = {0.f, 0.f, 0.f, 0.f, 0.f, 0.f, 0.f, 0.f};
  v8f acc[4];
#pragma unroll
  for (int nt = 0; nt < 4; ++nt) acc[nt] = vz;
#pragma unroll
  for (int kb = 0; kb < 4; ++kb) {
    v16bf a = frag_a_packed(ldsT[wave], lane, kb);
#pragma unroll
    for (int nt = 0; nt < 4; ++nt) {
      v16bf b = frag_b_packed(Wc2, 4, lane, kb, nt);
      acc[nt] = wmma_bf16(a, b, acc[nt]);
    }
  }

  const int ncol = lane & 15;
  const int mbase = (lane < 16) ? 0 : 8;
#pragma unroll
  for (int nt = 0; nt < 4; ++nt)
#pragma unroll
    for (int v = 0; v < 8; ++v) {
      int gr = r0 + mbase + v;
      if (gr < N_NODES) {
        float val = acc[nt][v] * dis[gr];
        size_t o = (size_t)gr * 64 + nt * 16 + ncol;
        hs2[o] = (__bf16)val;
        acc2[o] = val;
      }
    }
}

// ---- edge scatter-add, 64 features ----------------------------------------
__global__ void k_agg64(const int* __restrict__ src, const int* __restrict__ dst,
                        const __bf16* __restrict__ hs, float* __restrict__ acc) {
  long long t = (long long)blockIdx.x * blockDim.x + threadIdx.x;
  int e = (int)(t >> 6);
  int f = (int)(t & 63);
  if (e < N_EDGES) {
    int s = src[e], d = dst[e];
    atomicAdd(&acc[(size_t)d * 64 + f], (float)hs[(size_t)s * 64 + f]);
  }
}

// ---- conv2 epilogue + global mean-pool (segment sums) ---------------------
__global__ void k_pool(const float* __restrict__ acc2, const float* __restrict__ conv2_b,
                       const float* __restrict__ dis, const int* __restrict__ batch,
                       float* __restrict__ sums, float* __restrict__ cnt) {
  int t = blockIdx.x * blockDim.x + threadIdx.x;
  int node = t >> 6, f = t & 63;
  if (node < N_NODES) {
    float v = dis[node] * acc2[(size_t)node * 64 + f] + conv2_b[f];
    int g = batch[node];
    atomicAdd(&sums[g * 64 + f], v);
    if (f == 0) atomicAdd(&cnt[g], 1.0f);
  }
}

// ---- tiny head: pooled @ lin2_w + lin2_b ----------------------------------
__global__ void k_head(const float* __restrict__ sums, const float* __restrict__ cnt,
                       const float* __restrict__ lin2_w, const float* __restrict__ lin2_b,
                       float* __restrict__ out) {
  int t = blockIdx.x * blockDim.x + threadIdx.x;
  if (t < N_GR * N_CLS) {
    int g = t / N_CLS, c = t % N_CLS;
    float inv = 1.0f / fmaxf(cnt[g], 1.0f);
    float s = lin2_b[c];
#pragma unroll
    for (int f = 0; f < H2; ++f)
      s = fmaf(sums[g * 64 + f] * inv, lin2_w[f * N_CLS + c], s);
    out[t] = s;
  }
}

// ---------------------------------------------------------------------------

extern "C" void kernel_launch(void* const* d_in, const int* in_sizes, int n_in,
                              void* d_out, int out_size, void* d_ws, size_t ws_size,
                              hipStream_t stream) {
  const float* x       = (const float*)d_in[0];
  const int*   ei      = (const int*)d_in[1];   // [2, E]: src row then dst row
  const int*   batch   = (const int*)d_in[2];
  const float* drop_u  = (const float*)d_in[4];
  const float* lin1_w  = (const float*)d_in[5];
  const float* lin1_b  = (const float*)d_in[6];
  const float* conv1_w = (const float*)d_in[7];
  const float* conv1_b = (const float*)d_in[8];
  const float* conv2_w = (const float*)d_in[9];
  const float* conv2_b = (const float*)d_in[10];
  const float* lin2_w  = (const float*)d_in[11];
  const float* lin2_b  = (const float*)d_in[12];
  float* out = (float*)d_out;

  const int* e_src = ei;
  const int* e_dst = ei + N_EDGES;

  // workspace carve-up (256B aligned)
  char* ws = (char*)d_ws;
  size_t off = 0;
  auto take = [&](size_t bytes) {
    char* p = ws + off;
    off = (off + bytes + 255) & ~(size_t)255;
    return p;
  };
  float*  dis   = (float*)take((size_t)N_NODES * 4);           // deg then deg^-1/2
  __bf16* W1bf  = (__bf16*)take((size_t)D_IN * D_IN * 2);      // B-frag packed
  __bf16* Wc1bf = (__bf16*)take((size_t)D_IN * HID * 2);       // B-frag packed
  __bf16* Wc2bf = (__bf16*)take((size_t)HID * H2 * 2);         // B-frag packed
  __bf16* hs1   = (__bf16*)take((size_t)N_NODES * 128 * 2);
  float*  acc1  = (float*)take((size_t)N_NODES * 128 * 4);
  __bf16* hs2   = (__bf16*)take((size_t)N_NODES * 64 * 2);
  float*  acc2  = (float*)take((size_t)N_NODES * 64 * 4);
  float*  sums  = (float*)take((size_t)N_GR * H2 * 4);
  float*  cnt   = (float*)take((size_t)N_GR * 4);
  (void)ws_size; (void)in_sizes; (void)n_in; (void)out_size;

  const int T = 256;
  k_init<<<(N_NODES + T - 1) / T, T, 0, stream>>>(dis, sums, cnt);
  k_degree<<<(N_EDGES + T - 1) / T, T, 0, stream>>>(e_dst, dis);
  k_dis<<<(N_NODES + T - 1) / T, T, 0, stream>>>(dis);
  k_packw<<<(D_IN * D_IN + T - 1) / T, T, 0, stream>>>(lin1_w, conv1_w, conv2_w,
                                                       W1bf, Wc1bf, Wc2bf);
  k_gemm1<<<(N_NODES + 63) / 64, 128, 0, stream>>>(x, lin1_b, W1bf, Wc1bf, dis,
                                                   hs1, acc1);
  k_agg128<<<(int)(((long long)N_EDGES * 128 + T - 1) / T), T, 0, stream>>>(
      e_src, e_dst, hs1, acc1);
  k_post1_gemm2<<<(N_NODES + 63) / 64, 128, 0, stream>>>(acc1, conv1_b, drop_u,
                                                         Wc2bf, dis, hs2, acc2);
  k_agg64<<<(int)(((long long)N_EDGES * 64 + T - 1) / T), T, 0, stream>>>(
      e_src, e_dst, hs2, acc2);
  k_pool<<<(N_NODES * 64 + T - 1) / T, T, 0, stream>>>(acc2, conv2_b, dis, batch,
                                                       sums, cnt);
  k_head<<<(N_GR * N_CLS + T - 1) / T, T, 0, stream>>>(sums, cnt, lin2_w, lin2_b, out);
}
